// multi_head_attention_50148038148551
// MI455X (gfx1250) — matrix operand order repaired and checked by reading, not compiled, not run on hardware
//
#include <hip/hip_runtime.h>
#include <hip/hip_bf16.h>

// CDNA5 / gfx1250 multi-head attention.
// Matmuls: v_wmma_f32_16x16x32_f16. Attention staging: global_load_async_to_lds_b128
// (ASYNCcnt, double-buffered, NT hint on the 256MB bias stream).
// Softmax row reductions: DPP16 butterflies (no LDS traffic).

typedef __attribute__((ext_vector_type(16))) _Float16 v16h;
typedef __attribute__((ext_vector_type(8)))  float    v8f;

union AF { v16h h; unsigned u[8]; _Float16 f[16]; uint4 q[2]; };
union H2 { _Float16 h[2]; unsigned u; };

#define NHEAD 16
#define DMODEL 1024
#define MSEQ 1024
#define BATCH 4
#define DHEAD 64

// ---- gfx1250 async copy helpers ----
__device__ __forceinline__ void async_b128(void* ldsdst, const void* gsrc) {
    unsigned l = (unsigned)(uintptr_t)ldsdst;   // low 32 bits of flat == LDS offset
    asm volatile("global_load_async_to_lds_b128 %0, %1, off"
                 :: "v"(l), "v"(gsrc) : "memory");
}
__device__ __forceinline__ void async_b128_nt(void* ldsdst, const void* gsrc) {
    unsigned l = (unsigned)(uintptr_t)ldsdst;
    asm volatile("global_load_async_to_lds_b128 %0, %1, off th:TH_LOAD_NT"
                 :: "v"(l), "v"(gsrc) : "memory");
}
__device__ __forceinline__ void wait_async_le8() {
    asm volatile("s_wait_asynccnt 0x8" ::: "memory");
}
__device__ __forceinline__ void wait_async_0() {
    asm volatile("s_wait_asynccnt 0x0" ::: "memory");
}

// ---- DPP16 butterfly reductions over 16-lane groups (pure VALU) ----
template<int CTRL>
__device__ __forceinline__ float dpp_mv(float x) {
    return __int_as_float(__builtin_amdgcn_mov_dpp(__float_as_int(x), CTRL, 0xf, 0xf, true));
}
__device__ __forceinline__ float redmax16(float x) {
    x = fmaxf(x, dpp_mv<0xB1>(x));    // quad_perm [1,0,3,2]  (xor 1)
    x = fmaxf(x, dpp_mv<0x4E>(x));    // quad_perm [2,3,0,1]  (xor 2)
    x = fmaxf(x, dpp_mv<0x141>(x));   // row_half_mirror
    x = fmaxf(x, dpp_mv<0x140>(x));   // row_mirror
    return x;
}
__device__ __forceinline__ float redsum16(float x) {
    x += dpp_mv<0xB1>(x);
    x += dpp_mv<0x4E>(x);
    x += dpp_mv<0x141>(x);
    x += dpp_mv<0x140>(x);
    return x;
}

// ---------------------------------------------------------------------------
// Tiled GEMM: C[4096 x 1024] = A[4096 x 1024] * W[1024 x 1024] + bias
// 128 threads (4 waves); 64x64 tile; wave w: 4 Mx16 subtiles of 16-wide N col.
// LDS tiles stride 40 halves (80B: 16B-aligned rows, conflict-free).
// ---------------------------------------------------------------------------
#define TST 40
template<bool A_F16, bool HEADSPLIT>
__global__ __launch_bounds__(128)
void gemm128(const void* __restrict__ Ain, const float* __restrict__ W,
             const float* __restrict__ bias, void* __restrict__ Cout,
             float out_scale)
{
    const int tid  = threadIdx.x;
    const int lane = tid & 31;
    const int wv   = tid >> 5;
    const int m16  = lane & 15;
    const int hf   = lane >> 4;
    const int nbase = blockIdx.x * 64;
    const int mbase = blockIdx.y * 64;

    __shared__ __align__(16) _Float16 At[64 * TST];   // [m][k] 64x32
    __shared__ __align__(16) _Float16 Bt[64 * TST];   // W tile transposed: [n][k]

    v8f acc[4] = {};

    for (int kk = 0; kk < DMODEL; kk += 32) {
        __syncthreads();
        {
            const int row = tid >> 1;
            const int kof = (tid & 1) * 16;
            if (A_F16) {
                const _Float16* A = (const _Float16*)Ain;
                const uint4* src = (const uint4*)(A + (size_t)(mbase + row) * DMODEL + kk + kof);
                *(uint4*)&At[row * TST + kof]     = src[0];
                *(uint4*)&At[row * TST + kof + 8] = src[1];
            } else {
                const float* A = (const float*)Ain;
                const float4* src = (const float4*)(A + (size_t)(mbase + row) * DMODEL + kk + kof);
                #pragma unroll
                for (int j4 = 0; j4 < 4; ++j4) {
                    const float4 x = src[j4];
                    H2 p0, p1;
                    p0.h[0] = (_Float16)x.x; p0.h[1] = (_Float16)x.y;
                    p1.h[0] = (_Float16)x.z; p1.h[1] = (_Float16)x.w;
                    uint2 pk; pk.x = p0.u; pk.y = p1.u;
                    *(uint2*)&At[row * TST + kof + 4 * j4] = pk;
                }
            }
            const int kr  = tid >> 2;
            const int nc0 = (tid & 3) * 16;
            const float4* ws = (const float4*)(W + (size_t)(kk + kr) * DMODEL + nbase + nc0);
            #pragma unroll
            for (int j4 = 0; j4 < 4; ++j4) {
                const float4 x = ws[j4];
                Bt[(nc0 + 4 * j4 + 0) * TST + kr] = (_Float16)x.x;
                Bt[(nc0 + 4 * j4 + 1) * TST + kr] = (_Float16)x.y;
                Bt[(nc0 + 4 * j4 + 2) * TST + kr] = (_Float16)x.z;
                Bt[(nc0 + 4 * j4 + 3) * TST + kr] = (_Float16)x.w;
            }
        }
        __syncthreads();

        AF bf;   // B frag: n = lane%16, k = (lane/16)*16 + i  (16 contiguous halves)
        bf.q[0] = *(const uint4*)&Bt[(wv * 16 + m16) * TST + hf * 16];
        bf.q[1] = *(const uint4*)&Bt[(wv * 16 + m16) * TST + hf * 16 + 8];

        #pragma unroll
        for (int mt = 0; mt < 4; ++mt) {
            AF af;   // A frag: k = hf*16 + i (16 contiguous halves)
            af.q[0] = *(const uint4*)&At[(mt * 16 + m16) * TST + hf * 16];
            af.q[1] = *(const uint4*)&At[(mt * 16 + m16) * TST + hf * 16 + 8];
            acc[mt] = __builtin_amdgcn_wmma_f32_16x16x32_f16(
                false, af.h, false, bf.h, (short)0, acc[mt], false, false);
        }
    }

    const int gn = nbase + wv * 16 + m16;
    #pragma unroll
    for (int mt = 0; mt < 4; ++mt) {
        #pragma unroll
        for (int v = 0; v < 8; ++v) {
            const int gm = mbase + mt * 16 + v + 8 * hf;
            const float val = (acc[mt][v] + bias[gn]) * out_scale;
            if (HEADSPLIT) {
                _Float16* C = (_Float16*)Cout;
                const int bb = gm >> 10, mm = gm & 1023;
                const int head = gn >> 6, hk = gn & 63;
                C[(((size_t)(bb * NHEAD + head)) * MSEQ + mm) * DHEAD + hk] = (_Float16)val;
            } else {
                float* C = (float*)Cout;
                C[(size_t)gm * DMODEL + gn] = val;
            }
        }
    }
}

// ---------------------------------------------------------------------------
// Flash attention: block = (b, head, 64 q rows); 4 waves x 16 q rows.
// K/V/bias tiles double-buffered in LDS via global_load_async_to_lds_b128.
// ---------------------------------------------------------------------------
__global__ __launch_bounds__(128)
void attn128(const _Float16* __restrict__ Qh, const _Float16* __restrict__ Kh,
             const _Float16* __restrict__ Vh, const float* __restrict__ bias,
             _Float16* __restrict__ Ch)
{
    const int tid  = threadIdx.x;
    const int lane = tid & 31;
    const int wv   = tid >> 5;
    const int m16  = lane & 15;
    const int hf   = lane >> 4;
    const int qt = blockIdx.x, h = blockIdx.y, b = blockIdx.z;
    const size_t headbase = ((size_t)(b * NHEAD + h)) * MSEQ * DHEAD;
    const int gq = qt * 64 + wv * 16;

    __shared__ __align__(16) _Float16 Kt[2][32 * 64];    // [key][dk] row-major
    __shared__ __align__(16) _Float16 Vt[2][32 * 64];    // [key][vdim] row-major
    __shared__ __align__(16) float    Bs[2][64 * 32];    // bias tile [qrow][key]
    __shared__ __align__(16) _Float16 Pb[4][16 * TST];   // per-wave P transpose

    const _Float16* kgbase = Kh + headbase;
    const _Float16* vgbase = Vh + headbase;
    const float*    bgbase = bias + (((size_t)(b * NHEAD + h)) * MSEQ + (size_t)qt * 64) * MSEQ;

    auto issue = [&](int kb, int buf) {
        const size_t K0 = (size_t)kb * 32;
        #pragma unroll
        for (int t = 0; t < 2; ++t) {              // K, V: 4KB contiguous each
            const int ci = tid + t * 128;          // 16B chunk id, 0..255
            async_b128(&Kt[buf][ci * 8], kgbase + K0 * 64 + ci * 8);
            async_b128(&Vt[buf][ci * 8], vgbase + K0 * 64 + ci * 8);
        }
        #pragma unroll
        for (int t = 0; t < 4; ++t) {              // bias: 64 rows x 128B
            const int ci = tid + t * 128;          // 0..511
            const int row = ci >> 3, c4 = ci & 7;
            async_b128_nt(&Bs[buf][row * 32 + c4 * 4],
                          bgbase + (size_t)row * MSEQ + K0 + c4 * 4);
        }
    };

    // preload Q fragments (dk 0..31, 32..63); softmax scale folded at projection
    AF qf[2];
    {
        const _Float16* qp = Qh + headbase + (size_t)(gq + m16) * DHEAD;
        #pragma unroll
        for (int c = 0; c < 2; ++c) {
            qf[c].q[0] = *(const uint4*)&qp[c * 32 + hf * 16];
            qf[c].q[1] = *(const uint4*)&qp[c * 32 + hf * 16 + 8];
        }
    }

    float m_run[8], l_run[8];
    v8f o[4] = {};
    #pragma unroll
    for (int v = 0; v < 8; ++v) { m_run[v] = -1e30f; l_run[v] = 0.f; }

    const int NB = MSEQ / 32;
    issue(0, 0);

    for (int kb = 0; kb < NB; ++kb) {
        const int cur = kb & 1;
        __syncthreads();                      // all waves done reading buf cur^1
        if (kb + 1 < NB) { issue(kb + 1, cur ^ 1); wait_async_le8(); }
        else             { wait_async_0(); }
        __syncthreads();                      // buf cur visible to all waves

        // ---- logits for two 16-key subtiles ----
        float s[2][8];
        #pragma unroll
        for (int sb = 0; sb < 2; ++sb) {
            AF bk0, bk1;   // B = K^T: n = key (lane%16), k = dk
            bk0.q[0] = *(const uint4*)&Kt[cur][(sb * 16 + m16) * 64 + hf * 16];
            bk0.q[1] = *(const uint4*)&Kt[cur][(sb * 16 + m16) * 64 + hf * 16 + 8];
            bk1.q[0] = *(const uint4*)&Kt[cur][(sb * 16 + m16) * 64 + 32 + hf * 16];
            bk1.q[1] = *(const uint4*)&Kt[cur][(sb * 16 + m16) * 64 + 32 + hf * 16 + 8];
            v8f c = {};
            c = __builtin_amdgcn_wmma_f32_16x16x32_f16(false, qf[0].h, false, bk0.h, (short)0, c, false, false);
            c = __builtin_amdgcn_wmma_f32_16x16x32_f16(false, qf[1].h, false, bk1.h, (short)0, c, false, false);
            #pragma unroll
            for (int v = 0; v < 8; ++v)
                s[sb][v] = c[v] + Bs[cur][(wv * 16 + v + 8 * hf) * 32 + sb * 16 + m16];
        }

        // ---- online softmax (rows live in 16-lane half groups; DPP reductions) ----
        #pragma unroll
        for (int v = 0; v < 8; ++v) {
            const float tmax = redmax16(fmaxf(s[0][v], s[1][v]));
            const float mnew  = fmaxf(m_run[v], tmax);
            const float alpha = __expf(m_run[v] - mnew);
            const float p0 = __expf(s[0][v] - mnew);
            const float p1 = __expf(s[1][v] - mnew);
            const float rs = redsum16(p0 + p1);
            l_run[v] = l_run[v] * alpha + rs;
            m_run[v] = mnew;
            #pragma unroll
            for (int t = 0; t < 4; ++t) o[t][v] *= alpha;
            const int row = v + 8 * hf;
            Pb[wv][row * TST + m16]      = (_Float16)p0;
            Pb[wv][row * TST + 16 + m16] = (_Float16)p1;
        }
        __builtin_amdgcn_wave_barrier();

        // ---- P in A layout, ctx += P * V ----
        AF pf;
        pf.q[0] = *(const uint4*)&Pb[wv][m16 * TST + hf * 16];
        pf.q[1] = *(const uint4*)&Pb[wv][m16 * TST + hf * 16 + 8];
        #pragma unroll
        for (int t = 0; t < 4; ++t) {
            AF bv;   // B = V: n = vdim (lane%16), k = key; per-element from row-major tile
            #pragma unroll
            for (int i = 0; i < 16; ++i)
                bv.f[i] = Vt[cur][(hf * 16 + i) * 64 + t * 16 + m16];
            o[t] = __builtin_amdgcn_wmma_f32_16x16x32_f16(
                false, pf.h, false, bv.h, (short)0, o[t], false, false);
        }
    }

    // ---- normalize and write context [b, m, head*64 + c] as f16 ----
    #pragma unroll
    for (int v = 0; v < 8; ++v) {
        const float inv = 1.0f / l_run[v];
        const int row = gq + v + 8 * hf;
        _Float16* cp = Ch + ((size_t)b * MSEQ + row) * DMODEL + h * DHEAD;
        #pragma unroll
        for (int t = 0; t < 4; ++t)
            cp[t * 16 + m16] = (_Float16)(o[t][v] * inv);
    }
}

// ---------------------------------------------------------------------------
extern "C" void kernel_launch(void* const* d_in, const int* in_sizes, int n_in,
                              void* d_out, int out_size, void* d_ws, size_t ws_size,
                              hipStream_t stream) {
    const float* queries = (const float*)d_in[0];
    const float* keys    = (const float*)d_in[1];
    const float* values  = (const float*)d_in[2];
    const float* abias   = (const float*)d_in[3];
    const float* WQ = (const float*)d_in[4];  const float* bQ = (const float*)d_in[5];
    const float* WK = (const float*)d_in[6];  const float* bK = (const float*)d_in[7];
    const float* WV = (const float*)d_in[8];  const float* bV = (const float*)d_in[9];
    const float* WO = (const float*)d_in[10]; const float* bO = (const float*)d_in[11];
    float* out = (float*)d_out;

    const size_t HELEMS = (size_t)BATCH * NHEAD * MSEQ * DHEAD;  // 4,194,304
    _Float16* Qh = (_Float16*)d_ws;
    _Float16* Kh = Qh + HELEMS;
    _Float16* Vh = Kh + HELEMS;
    _Float16* Ch = Vh + HELEMS;

    const dim3 gblk(128), gemmGrid(DMODEL / 64, (BATCH * MSEQ) / 64);
    const float qscale = 0.125f;  // 1/sqrt(64)

    gemm128<false, true><<<gemmGrid, gblk, 0, stream>>>(queries, WQ, bQ, Qh, qscale);
    gemm128<false, true><<<gemmGrid, gblk, 0, stream>>>(keys,    WK, bK, Kh, 1.0f);
    gemm128<false, true><<<gemmGrid, gblk, 0, stream>>>(values,  WV, bV, Vh, 1.0f);

    attn128<<<dim3(MSEQ / 64, NHEAD, BATCH), gblk, 0, stream>>>(Qh, Kh, Vh, abias, Ch);

    gemm128<true, false><<<gemmGrid, gblk, 0, stream>>>(Ch, WO, bO, out, 1.0f);
}
